// GPSA1D_20444044329743
// MI455X (gfx1250) — matrix operand order repaired and checked by reading, not compiled, not run on hardware
//
#include <hip/hip_runtime.h>
#include <hip/hip_bf16.h>
#include <math.h>

typedef __bf16 bf16;
typedef __attribute__((ext_vector_type(16))) __bf16 v16bf;
typedef __attribute__((ext_vector_type(8)))  float  v8f;
typedef __attribute__((ext_vector_type(4)))  int    v4i;

#define DIMC   256
#define NPATCH 2048
#define NHEADS 8
#define NBATCH 2
#define DH     32
#define NROWS  (NBATCH * NPATCH)   // 4096
#define NCHUNK (NPATCH / 32)       // 64

#define V8F_ZERO {0.f,0.f,0.f,0.f,0.f,0.f,0.f,0.f}
#define LOG2E 1.4426950408889634f
#define QSC   (0.1767766952966369f * LOG2E)  // 32^-0.5 * log2e, folded into Q

// ---- optional gfx1250 async global->LDS copy path ----
#if defined(__has_builtin)
#  if __has_builtin(__builtin_amdgcn_global_load_async_to_lds_b128)
#    define HAVE_ASYNC_COPY 1
#  endif
#endif
#ifndef HAVE_ASYNC_COPY
#  define HAVE_ASYNC_COPY 0
#endif

#if HAVE_ASYNC_COPY
__device__ __forceinline__ void async_copy16(const bf16* gsrc, bf16* ldst) {
  __builtin_amdgcn_global_load_async_to_lds_b128(
      (__attribute__((address_space(1))) v4i*)gsrc,
      (__attribute__((address_space(3))) v4i*)ldst,
      0, 0);
}
#  if defined(__has_builtin) && __has_builtin(__builtin_amdgcn_s_wait_asynccnt)
#    define ASYNC_WAIT(n) __builtin_amdgcn_s_wait_asynccnt(n)
#  else
#    define ASYNC_WAIT(n) asm volatile("s_wait_asynccnt %0" :: "i"(n) : "memory")
#  endif
#else
#  define ASYNC_WAIT(n) do {} while (0)
#endif

__device__ __forceinline__ v8f wmma_bf16(v16bf a, v16bf b, v8f c) {
  return __builtin_amdgcn_wmma_f32_16x16x32_bf16(false, a, false, b, (short)0, c,
                                                 false, false);
}

// ---- A fragment (16x32 bf16), row-major source, leading dim ld (elements).
__device__ __forceinline__ v16bf load_fragA(const bf16* p, int ld) {
  int lane = threadIdx.x & 31;
  int rc = lane & 15;
  int kb = (lane >> 4) << 3;  // 0 or 8
  const bf16* q = p + rc * ld + kb;
  union { uint4 u[2]; v16bf v; } t;
  t.u[0] = *(const uint4*)(q);
  t.u[1] = *(const uint4*)(q + 16);
  return t.v;
}

// ---- B fragment (32x16 bf16, B[k][n]); source stores B^T row-major.
__device__ __forceinline__ v16bf load_fragB(const bf16* p, int ld) {
  int lane = threadIdx.x & 31;
  int nc = lane & 15;
  int kb = (lane >> 4) << 3;  // 0 or 8
  const bf16* q = p + nc * ld + kb;
  union { uint4 u[2]; v16bf v; } t;
  t.u[0] = *(const uint4*)(q);
  t.u[1] = *(const uint4*)(q + 16);
  return t.v;
}

__device__ __forceinline__ float hmax16(float v) {
  v = fmaxf(v, __shfl_xor(v, 1, 32));
  v = fmaxf(v, __shfl_xor(v, 2, 32));
  v = fmaxf(v, __shfl_xor(v, 4, 32));
  v = fmaxf(v, __shfl_xor(v, 8, 32));
  return v;
}
__device__ __forceinline__ float hsum16(float v) {
  v += __shfl_xor(v, 1, 32);
  v += __shfl_xor(v, 2, 32);
  v += __shfl_xor(v, 4, 32);
  v += __shfl_xor(v, 8, 32);
  return v;
}

// ------------------------- prep kernels -------------------------
__global__ void cvt_bf16_kernel(const float* __restrict__ x, bf16* __restrict__ y, int n) {
  int i = blockIdx.x * blockDim.x + threadIdx.x;
  if (i < n) y[i] = (bf16)x[i];
}

__global__ void pack_w_qkv_kernel(const float* __restrict__ Wqk,
                                  const float* __restrict__ Wv,
                                  bf16* __restrict__ Wt) {
  int j = blockIdx.x;   // 0..767
  int k = threadIdx.x;  // 0..255
  float v = (j < 512) ? Wqk[k * 512 + j] : Wv[k * 256 + (j - 512)];
  Wt[j * 256 + k] = (bf16)v;
}

__global__ void pack_w_proj_kernel(const float* __restrict__ Wp, bf16* __restrict__ Wt) {
  int j = blockIdx.x;
  int k = threadIdx.x;
  Wt[j * 256 + k] = (bf16)Wp[k * 256 + j];
}

// ------------------------- QKV GEMM -------------------------
__global__ void __launch_bounds__(256) qkv_gemm_kernel(
    const bf16* __restrict__ xb, const bf16* __restrict__ Wt,
    bf16* __restrict__ Qb, bf16* __restrict__ Kb, bf16* __restrict__ Vt) {
  int wave = threadIdx.x >> 5;
  int lane = threadIdx.x & 31;
  int tile = blockIdx.x * 8 + wave;  // 256 row-tiles * 48 col-tiles
  int rt = tile / 48, ct = tile - rt * 48;
  int m0 = rt * 16, j0 = ct * 16;

  v8f acc = V8F_ZERO;
#pragma unroll 2
  for (int k0 = 0; k0 < DIMC; k0 += 32) {
    v16bf a = load_fragA(xb + (size_t)m0 * DIMC + k0, DIMC);
    v16bf b = load_fragB(Wt + (size_t)j0 * DIMC + k0, DIMC);
    acc = wmma_bf16(a, b, acc);
  }

  int nc = lane & 15, hi = lane >> 4;
  int j = j0 + nc;
  int mbase = m0 + 8 * hi;
  int sel = j >> 8;  // 0:Q 1:K 2:V (uniform per tile)
  int c = j & 255;
  int h = c >> 5, dh = c & 31;
  int bb = mbase >> 11, nrow0 = mbase & 2047;

  if (sel < 2) {
    float qs = (sel == 0) ? QSC : 1.f;  // fold softmax scale (log2 domain) into Q
    bf16* dst = (sel == 0 ? Qb : Kb) +
                (((size_t)(bb * NHEADS + h) * NPATCH + nrow0) * DH + dh);
#pragma unroll
    for (int r = 0; r < 8; r++) dst[(size_t)r * DH] = (bf16)(acc[r] * qs);
  } else {
    bf16* dst = Vt + ((size_t)(bb * NHEADS + h) * DH + dh) * NPATCH + nrow0;
    union { uint4 u; bf16 hh[8]; } pk;
#pragma unroll
    for (int r = 0; r < 8; r++) pk.hh[r] = (bf16)acc[r];
    *(uint4*)dst = pk.u;
  }
}

// ------------------------- fused dual flash attention -------------------------
// out_head = (1-g)*softmax(QK^T*scale)@V + g*softmax(Pos)@V
// 8 waves/block share one (b,h); K/V chunks double-buffered in LDS via async copy.
// Chunk key order is interleaved: K-tile0 = even keys, K-tile1 = odd keys, so the
// content P store packs (p_even, p_odd) as one dword while V staging stays identity.
__global__ void __launch_bounds__(256) attn_kernel(
    const bf16* __restrict__ Qb, const bf16* __restrict__ Kb,
    const bf16* __restrict__ Vt, const float* __restrict__ Wpos,
    const float* __restrict__ bpos, const float* __restrict__ gating,
    bf16* __restrict__ ctx) {
  __shared__ __align__(16) bf16 ldsK[2][32 * 32];  // 4 KB double-buffered K chunk
  __shared__ __align__(16) bf16 ldsV[2][32 * 32];  // 4 KB double-buffered V chunk
  __shared__ __align__(16) bf16 pst[8][16 * 32];   // 8 KB per-wave P staging

  int wave = threadIdx.x >> 5;
  int lane = threadIdx.x & 31;
  int nc = lane & 15, hi = lane >> 4;
  int tile = blockIdx.x * 8 + wave;  // 2048 M-tiles; 8 per block, same bh
  int bh = tile >> 7;
  int m0 = (tile & 127) << 4;
  int b = bh >> 3, h = bh & 7;

  const bf16* Q = Qb + (size_t)bh * NPATCH * DH;
  const bf16* K = Kb + (size_t)bh * NPATCH * DH;
  const bf16* V = Vt + (size_t)bh * DH * NPATCH;

  float w0l = Wpos[h] * LOG2E, w1l = Wpos[NHEADS + h] * LOG2E;
  float bpl = bpos[h] * LOG2E;
  float g = 1.f / (1.f + __expf(-gating[h]));

  v16bf aQ = load_fragA(Q + (size_t)m0 * DH, DH);  // Q pre-scaled by QSC

  v8f oc0 = V8F_ZERO, oc1 = V8F_ZERO;  // content accumulators
  v8f op0 = V8F_ZERO, op1 = V8F_ZERO;  // positional accumulators
  float mc[8], lc[8];
#pragma unroll
  for (int r = 0; r < 8; r++) { mc[r] = -INFINITY; lc[r] = 0.f; }

  // ---- positional setup: A-layout per-lane row, exact analytic max ----
  int kbA = (lane >> 4) << 3;              // A-frag K base: 0 or 8
  float mrA = (float)(m0 + (lane & 15));   // A-frag row
  float f0 = (w0l + w1l) * mrA + bpl;               // pos logit at n=0
  float fe = (w1l - w0l) * (2047.f - mrA) + bpl;    // pos logit at n=N-1
  float mpA = fmaxf(bpl, fmaxf(f0, fe));            // exact global max (kink at n=m)
  float bm = bpl - mpA;                             // folded bias
  float lpA = 0.f;                                  // per-lane partial row sum

  // ---- staging roles: threads 0..127 copy K chunk, 128..255 copy V chunk ----
  int t = threadIdx.x;
  int j = (t & 127) >> 2;          // LDS row 0..31
  int ccol = (t & 3) << 3;         // 8-element column offset
  bool doK = (t < 128);
  int growK = (j < 16) ? (2 * j) : (2 * j - 31);  // even keys -> tile0, odd -> tile1
  const bf16* gbase = doK ? (K + (size_t)growK * DH + ccol)
                          : (V + (size_t)j * NPATCH + ccol);
  size_t gstep = doK ? (size_t)DH : (size_t)1;  // advance per unit of n0
  bf16* lb[2];
  lb[0] = doK ? (ldsK[0] + (size_t)t * 8) : (ldsV[0] + (size_t)(t - 128) * 8);
  lb[1] = doK ? (ldsK[1] + (size_t)t * 8) : (ldsV[1] + (size_t)(t - 128) * 8);

  auto stage = [&](int buf, int n0s) {
    const bf16* gsrc = gbase + (size_t)n0s * gstep;
#if HAVE_ASYNC_COPY
    async_copy16(gsrc, lb[buf]);
#else
    *(uint4*)lb[buf] = *(const uint4*)gsrc;
#endif
  };

  bf16* P = pst[wave];
  stage(0, 0);

  for (int i = 0; i < NCHUNK; i++) {
    int n0 = i * 32;
    int cur = i & 1;
    if (i + 1 < NCHUNK) {
      stage(cur ^ 1, n0 + 32);  // prefetch next chunk (buffer read-safe: barrier below)
      ASYNC_WAIT(1);            // current chunk landed; next may stay in flight
    } else {
      ASYNC_WAIT(0);
    }
    __syncthreads();

    const bf16* cK = ldsK[cur];
    const bf16* cV = ldsV[cur];
    v16bf bK0 = load_fragB(cK, 32);           // even keys n0+2*nc
    v16bf bK1 = load_fragB(cK + 16 * 32, 32); // odd keys  n0+2*nc+1
    v16bf bV0 = load_fragB(cV, 32);           // d columns 0..15
    v16bf bV1 = load_fragB(cV + 16 * 32, 32); // d columns 16..31

    v8f z = V8F_ZERO;
    v8f s0 = wmma_bf16(aQ, bK0, z);
    v8f s1 = wmma_bf16(aQ, bK1, z);

    // ---- content: online softmax (log2 domain; scale already folded into Q) ----
#pragma unroll
    for (int r = 0; r < 8; r++) {
      float a0 = s0[r], a1 = s1[r];
      float rm = hmax16(fmaxf(a0, a1));
      float mn = fmaxf(mc[r], rm);
      float al = exp2f(mc[r] - mn);
      mc[r] = mn;
      float p0 = exp2f(a0 - mn), p1 = exp2f(a1 - mn);
      lc[r] = lc[r] * al + hsum16(p0 + p1);
      oc0[r] *= al; oc1[r] *= al;
      union { bf16 hh[2]; unsigned u; } pk;
      pk.hh[0] = (bf16)p0; pk.hh[1] = (bf16)p1;  // key columns (2nc, 2nc+1)
      ((unsigned*)P)[(r + 8 * hi) * 16 + nc] = pk.u;
    }
    asm volatile("s_wait_dscnt 0" ::: "memory");
    v16bf aPc = load_fragA(P, 32);
    oc0 = wmma_bf16(aPc, bV0, oc0);
    oc1 = wmma_bf16(aPc, bV1, oc1);

    // ---- positional: built directly in A-fragment registers, no LDS/shuffles ----
    float nb = (float)n0;
    v16bf aPp;
    float psum = 0.f;
#pragma unroll
    for (int e = 0; e < 16; e++) {
      int nl = kbA + ((e < 8) ? e : e + 8);  // A-layout K slot -> chunk-local key
      float d = mrA - (nb + (float)nl);
      float aa = fmaf(w0l, d, fmaf(w1l, fabsf(d), bm));
      float p = exp2f(aa);
      psum += p;
      aPp[e] = (bf16)p;
    }
    lpA += psum;
    op0 = wmma_bf16(aPp, bV0, op0);
    op1 = wmma_bf16(aPp, bV1, op1);

    __syncthreads();  // all reads of this chunk done before it is re-staged
  }

  // redistribute positional row sums: row m lives in lanes m and m+16 (A-layout)
  float lpT = lpA + __shfl_xor(lpA, 16, 32);

#pragma unroll
  for (int r = 0; r < 8; r++) {
    int m = m0 + r + 8 * hi;
    float lpr = __shfl(lpT, r + 8 * hi, 32);  // C-layout row's pos denominator
    float o0 = (1.f - g) * oc0[r] / lc[r] + g * op0[r] / lpr;
    float o1 = (1.f - g) * oc1[r] / lc[r] + g * op1[r] / lpr;
    size_t base = ((size_t)(b * NPATCH + m)) * DIMC + h * DH;
    ctx[base + nc]      = (bf16)o0;
    ctx[base + 16 + nc] = (bf16)o1;
  }
}

// ------------------------- output projection -------------------------
__global__ void __launch_bounds__(256) proj_gemm_kernel(
    const bf16* __restrict__ ctx, const bf16* __restrict__ Wt,
    const float* __restrict__ bias, float* __restrict__ out) {
  int wave = threadIdx.x >> 5;
  int lane = threadIdx.x & 31;
  int tile = blockIdx.x * 8 + wave;  // 256 row-tiles * 16 col-tiles
  int rt = tile >> 4, ct = tile & 15;
  int m0 = rt * 16, n0 = ct * 16;

  v8f acc = V8F_ZERO;
#pragma unroll 2
  for (int k0 = 0; k0 < DIMC; k0 += 32) {
    v16bf a = load_fragA(ctx + (size_t)m0 * DIMC + k0, DIMC);
    v16bf b = load_fragB(Wt + (size_t)n0 * DIMC + k0, DIMC);
    acc = wmma_bf16(a, b, acc);
  }

  int nc = lane & 15, hi = lane >> 4;
  int jj = n0 + nc;
  float bj = bias[jj];
#pragma unroll
  for (int r = 0; r < 8; r++) {
    out[(size_t)(m0 + r + 8 * hi) * DIMC + jj] = acc[r] + bj;
  }
}

// ------------------------- host launcher -------------------------
extern "C" void kernel_launch(void* const* d_in, const int* in_sizes, int n_in,
                              void* d_out, int out_size, void* d_ws, size_t ws_size,
                              hipStream_t stream) {
  const float* x    = (const float*)d_in[0];
  const float* Wqk  = (const float*)d_in[1];
  const float* Wv   = (const float*)d_in[2];
  const float* Wp   = (const float*)d_in[3];
  const float* bpj  = (const float*)d_in[4];
  const float* Wpos = (const float*)d_in[5];
  const float* bpos = (const float*)d_in[6];
  const float* gat  = (const float*)d_in[7];
  float* out = (float*)d_out;

  char* ws = (char*)d_ws;
  const size_t MB = (size_t)1 << 20;
  bf16* xb    = (bf16*)(ws + 0 * MB);
  bf16* WtQKV = (bf16*)(ws + 2 * MB);
  bf16* WtP   = (bf16*)(ws + 2 * MB + 512 * 1024);
  bf16* Qb    = (bf16*)(ws + 3 * MB);
  bf16* Kb    = (bf16*)(ws + 5 * MB);
  bf16* Vt    = (bf16*)(ws + 7 * MB);
  bf16* ctx   = (bf16*)(ws + 9 * MB);

  cvt_bf16_kernel<<<(NROWS * DIMC) / 256, 256, 0, stream>>>(x, xb, NROWS * DIMC);
  pack_w_qkv_kernel<<<768, 256, 0, stream>>>(Wqk, Wv, WtQKV);
  pack_w_proj_kernel<<<256, 256, 0, stream>>>(Wp, WtP);

  qkv_gemm_kernel<<<(256 * 48) / 8, 256, 0, stream>>>(xb, WtQKV, Qb, Kb, Vt);

  attn_kernel<<<(NBATCH * NHEADS * (NPATCH / 16)) / 8, 256, 0, stream>>>(
      Qb, Kb, Vt, Wpos, bpos, gat, ctx);

  proj_gemm_kernel<<<(256 * 16) / 8, 256, 0, stream>>>(ctx, WtP, bpj, out);
}
